// DualHeadGNN_43078521979220
// MI455X (gfx1250) — compile-verified, matrix operand order checked by reading, not run on hardware
//
#include <hip/hip_runtime.h>
#include <hip/hip_bf16.h>
#include <math.h>
#include <stdint.h>

// ---------- WMMA fragment plumbing (gfx1250, wave32) ----------
typedef __attribute__((ext_vector_type(16))) __bf16 v16bf;
typedef __attribute__((ext_vector_type(8)))  float  v8f;

struct U4 { unsigned int a, b, c, d; };
union Frag {
  v16bf v;
  unsigned int   u32[8];
  unsigned short u16[16];
  U4 q[2];
};

__device__ __forceinline__ unsigned short f2bf(float f) {
  unsigned int u = __float_as_uint(f);
  unsigned int r = u + 0x7FFFu + ((u >> 16) & 1u);
  return (unsigned short)(r >> 16);
}
__device__ __forceinline__ unsigned int f2bf2(float lo, float hi) {
  return (unsigned int)f2bf(lo) | ((unsigned int)f2bf(hi) << 16);
}
// 16-bit A/B fragment K index for element j in half 'h' (lane>=16 -> h=1)
__device__ __forceinline__ int kmap(int j, int h) {
  return (j < 8) ? (j + 8 * h) : (j + 8 + 8 * h);
}

// gfx1250 async global->LDS copy (ASYNCcnt-tracked, no VGPR round trip)
__device__ __forceinline__ void async_load_b128(unsigned lds_addr, const void* gaddr) {
  asm volatile("global_load_async_to_lds_b128 %0, %1, off"
               :: "v"(lds_addr), "v"(gaddr) : "memory");
}
__device__ __forceinline__ void wait_async0() {
  asm volatile("s_wait_asynccnt 0" ::: "memory");
}

#define NEGF -3.402823466e38f

// ---------- f32 -> packed bf16 pair conversion ----------
__global__ void cvt_bf16_kernel(const float* __restrict__ in, unsigned int* __restrict__ out, int npairs) {
  int i = blockIdx.x * blockDim.x + threadIdx.x;
  if (i >= npairs) return;
  out[i] = f2bf2(in[2 * i], in[2 * i + 1]);
}

// ---------- generic WMMA GEMM: Y = act(Xbf[Mr,K] @ Wbf[Nc,K]^T + bias (+res)) ----------
// 256 threads (8 waves), 64x64 tile; K mult of 32, Mr/Nc mult of 64. Output f32 or bf16.
__global__ void gemm_bf16_kernel(const unsigned short* __restrict__ X,
                                 const unsigned short* __restrict__ W,
                                 const float* __restrict__ bias, const float* __restrict__ res,
                                 float* __restrict__ Yf, unsigned short* __restrict__ Ybf,
                                 int Mr, int Nc, int K, int ldw, int relu) {
  __shared__ unsigned int Xs[64 * 16];   // 64x32 bf16
  __shared__ unsigned int Ws[64 * 16];
  const int tid  = threadIdx.x;
  const int lane = tid & 31, wv = tid >> 5;
  const int mBase = blockIdx.y * 64, nBase = blockIdx.x * 64;
  const int r = lane & 15, hlf = lane >> 4;
  const int tn = wv & 3, tm0 = wv >> 2;   // wave owns tiles (tm0,tn) and (tm0+2,tn)
  const int qr = tid >> 2, qc = tid & 3;  // staging: one b128 per matrix per thread
  v8f acc0 = {}; v8f acc1 = {};

  for (int kk = 0; kk < K; kk += 32) {
    async_load_b128((unsigned)(uintptr_t)&Xs[qr * 16 + qc * 4],
                    X + (size_t)(mBase + qr) * K + kk + qc * 8);
    async_load_b128((unsigned)(uintptr_t)&Ws[qr * 16 + qc * 4],
                    W + (size_t)(nBase + qr) * ldw + kk + qc * 8);
    wait_async0();
    __syncthreads();
    Frag bfr, a0, a1;
    bfr.q[0] = *(const U4*)&Ws[(tn * 16 + r) * 16 + 4 * hlf];
    bfr.q[1] = *(const U4*)&Ws[(tn * 16 + r) * 16 + 8 + 4 * hlf];
    a0.q[0]  = *(const U4*)&Xs[(tm0 * 16 + r) * 16 + 4 * hlf];
    a0.q[1]  = *(const U4*)&Xs[(tm0 * 16 + r) * 16 + 8 + 4 * hlf];
    a1.q[0]  = *(const U4*)&Xs[((tm0 + 2) * 16 + r) * 16 + 4 * hlf];
    a1.q[1]  = *(const U4*)&Xs[((tm0 + 2) * 16 + r) * 16 + 8 + 4 * hlf];
    acc0 = __builtin_amdgcn_wmma_f32_16x16x32_bf16(false, a0.v, false, bfr.v, (short)0, acc0, false, false);
    acc1 = __builtin_amdgcn_wmma_f32_16x16x32_bf16(false, a1.v, false, bfr.v, (short)0, acc1, false, false);
    __syncthreads();
  }
  const int gcol = nBase + tn * 16 + r;
  const float bv = bias ? bias[gcol] : 0.f;
#pragma unroll
  for (int v = 0; v < 8; ++v) {
    int grow0 = mBase + tm0 * 16 + v + 8 * hlf;
    int grow1 = grow0 + 32;
    float y0 = acc0[v] + bv;
    float y1 = acc1[v] + bv;
    if (res)  { y0 += res[(size_t)grow0 * Nc + gcol]; y1 += res[(size_t)grow1 * Nc + gcol]; }
    if (relu) { y0 = fmaxf(y0, 0.f); y1 = fmaxf(y1, 0.f); }
    if (Ybf) {
      Ybf[(size_t)grow0 * Nc + gcol] = f2bf(y0);
      Ybf[(size_t)grow1 * Nc + gcol] = f2bf(y1);
    } else {
      Yf[(size_t)grow0 * Nc + gcol] = y0;
      Yf[(size_t)grow1 * Nc + gcol] = y1;
    }
  }
}

// ---------- fused edge-biased attention, bf16 QKV in, bf16 O out ----------
__global__ void attn_kernel(const unsigned short* __restrict__ qkv, const float* __restrict__ comm,
                            const int* __restrict__ adj, const unsigned char* __restrict__ mask,
                            const float* __restrict__ edgeW, unsigned short* __restrict__ o_bf) {
  __shared__ float        S[16 * 512];     // 32KB scores
  __shared__ unsigned int Qs[16 * 16];     // 1KB  Q tile
  __shared__ unsigned int Pb[16 * 256];    // 16KB probs (bf16 pairs)
  __shared__ float        Oacc[16 * 32];   // 2KB

  const int bidx = blockIdx.x;
  const int qt = bidx & 31, h = (bidx >> 5) & 7, b = bidx >> 8;
  const int qbase = qt * 16;
  const int tid = threadIdx.x, lane = tid & 31, wv = tid >> 5;
  const int r = lane & 15, hlf = lane >> 4;
  const float scale = 0.17677669529663687f;  // 1/sqrt(32)

  // stage Q (16x32 bf16) — straight dword copy
  {
    int row = tid >> 4, cp = tid & 15;
    const unsigned int* q32 =
        (const unsigned int*)(qkv + (size_t)(b * 512 + qbase + row) * 768 + h * 32);
    Qs[row * 16 + cp] = q32[cp];
  }
  __syncthreads();

  Frag aq;
  aq.q[0] = *(const U4*)&Qs[r * 16 + 4 * hlf];
  aq.q[1] = *(const U4*)&Qs[r * 16 + 8 + 4 * hlf];
  const float ew = edgeW[h];

  // phase 1: S = (Q K^T)*scale + bias; 4 key tiles per wave
  for (int kt = wv * 4; kt < wv * 4 + 4; ++kt) {
    const int key = kt * 16 + r;
    const unsigned short* kp = qkv + (size_t)(b * 512 + key) * 768 + 256 + h * 32;
    Frag bk;
    bk.q[0] = *(const U4*)(kp + 8 * hlf);
    bk.q[1] = *(const U4*)(kp + 16 + 8 * hlf);
    v8f acc = {};
    acc = __builtin_amdgcn_wmma_f32_16x16x32_bf16(false, aq.v, false, bk.v, (short)0, acc, false, false);
#pragma unroll
    for (int v = 0; v < 8; ++v) {
      int ql = v + 8 * hlf;
      int kg = kt * 16 + r;
      int qg = qbase + ql;
      float s = acc[v] * scale;
      int   a  = adj[(size_t)(b * 512 + qg) * 512 + kg];
      float eb = comm[(size_t)(b * 512 + qg) * 512 + kg] * 0.01f * ew;
      s += (a > 0) ? eb : NEGF;
      if (!mask[b * 512 + kg]) s = NEGF;
      S[ql * 512 + kg] = s;
    }
  }
  __syncthreads();

  // phase 2: row softmax, 2 rows per wave, P -> bf16 pairs in LDS
  for (int rr = wv * 2; rr < wv * 2 + 2; ++rr) {
    float m = -INFINITY;
    for (int c = lane; c < 512; c += 32) m = fmaxf(m, S[rr * 512 + c]);
    for (int off = 16; off > 0; off >>= 1) m = fmaxf(m, __shfl_xor(m, off, 32));
    float sum = 0.f;
    for (int c = lane; c < 512; c += 32) {
      float e = __expf(S[rr * 512 + c] - m);
      S[rr * 512 + c] = e;
      sum += e;
    }
    for (int off = 16; off > 0; off >>= 1) sum += __shfl_xor(sum, off, 32);
    float inv = 1.0f / sum;
    for (int pi = lane; pi < 256; pi += 32)
      Pb[rr * 256 + pi] = f2bf2(S[rr * 512 + 2 * pi] * inv, S[rr * 512 + 2 * pi + 1] * inv);
  }
  __syncthreads();

  // phase 3: O = P V, K split across waves, LDS float reduce
  Oacc[tid] = 0.f; Oacc[tid + 256] = 0.f;
  __syncthreads();
  v8f ac0 = {}; v8f ac1 = {};
  for (int ci = 0; ci < 2; ++ci) {
    const int kk = (wv + ci * 8) * 32;
    Frag ap;
    ap.q[0] = *(const U4*)&Pb[r * 256 + kk / 2 + 4 * hlf];
    ap.q[1] = *(const U4*)&Pb[r * 256 + kk / 2 + 8 + 4 * hlf];
#pragma unroll
    for (int dt = 0; dt < 2; ++dt) {
      const unsigned short* vbase =
          qkv + (size_t)b * 512 * 768 + 512 + h * 32 + dt * 16 + r;
      Frag bv;
#pragma unroll
      for (int j = 0; j < 16; ++j)
        bv.u16[j] = vbase[(size_t)(kk + kmap(j, hlf)) * 768];
      if (dt == 0) ac0 = __builtin_amdgcn_wmma_f32_16x16x32_bf16(false, ap.v, false, bv.v, (short)0, ac0, false, false);
      else         ac1 = __builtin_amdgcn_wmma_f32_16x16x32_bf16(false, ap.v, false, bv.v, (short)0, ac1, false, false);
    }
  }
#pragma unroll
  for (int v = 0; v < 8; ++v) {
    int M = v + 8 * hlf;
    atomicAdd(&Oacc[M * 32 + r], ac0[v]);
    atomicAdd(&Oacc[M * 32 + 16 + r], ac1[v]);
  }
  __syncthreads();
  {
    int row = tid >> 4, dp = tid & 15;
    unsigned int* o32 = (unsigned int*)o_bf;
    o32[((size_t)(b * 512 + qbase + row) * 256 + h * 32) / 2 + dp] =
        f2bf2(Oacc[row * 32 + 2 * dp], Oacc[row * 32 + 2 * dp + 1]);
  }
}

// ---------- LayerNorm over D=256, bf16 output ----------
__global__ void ln256_kernel(const float* __restrict__ x, const float* __restrict__ g,
                             const float* __restrict__ bb, unsigned short* __restrict__ y) {
  __shared__ float red[256];
  const int row = blockIdx.x, t = threadIdx.x;
  float v = x[(size_t)row * 256 + t];
  red[t] = v; __syncthreads();
  for (int s = 128; s > 0; s >>= 1) { if (t < s) red[t] += red[t + s]; __syncthreads(); }
  float mu = red[0] * (1.0f / 256.0f);
  __syncthreads();
  float d = v - mu;
  red[t] = d * d; __syncthreads();
  for (int s = 128; s > 0; s >>= 1) { if (t < s) red[t] += red[t + s]; __syncthreads(); }
  float var = red[0] * (1.0f / 256.0f);
  y[(size_t)row * 256 + t] = f2bf(d * rsqrtf(var + 1e-5f) * g[t] + bb[t]);
}

// ---------- node feature LN (F=8) ----------
__global__ void nfln_kernel(const float* __restrict__ ts, const float* __restrict__ cc,
                            const float* __restrict__ g, const float* __restrict__ bb,
                            float* __restrict__ out, int rows) {
  int row = blockIdx.x * blockDim.x + threadIdx.x;
  if (row >= rows) return;
  float f[8];
#pragma unroll
  for (int i = 0; i < 7; ++i) f[i] = ts[row * 7 + i];
  f[7] = cc[row] * 0.01f;
  float mu = 0.f;
#pragma unroll
  for (int i = 0; i < 8; ++i) mu += f[i];
  mu *= 0.125f;
  float var = 0.f;
#pragma unroll
  for (int i = 0; i < 8; ++i) { float d = f[i] - mu; var += d * d; }
  var *= 0.125f;
  float rs = rsqrtf(var + 1e-5f);
#pragma unroll
  for (int i = 0; i < 8; ++i) out[row * 8 + i] = (f[i] - mu) * rs * g[i] + bb[i];
}

// ---------- x = nfln @ node_W^T + node_b + pos_emb (K=8, VALU) ----------
__global__ void nodeproj_kernel(const float* __restrict__ nfln, const float* __restrict__ W,
                                const float* __restrict__ bias, const float* __restrict__ pos,
                                float* __restrict__ x) {
  int idx = blockIdx.x * 256 + threadIdx.x;   // B*N*D
  int row = idx >> 8, d = idx & 255;
  const float* f = nfln + row * 8;
  const float* w = W + d * 8;
  float acc = bias[d];
#pragma unroll
  for (int i = 0; i < 8; ++i) acc += f[i] * w[i];
  x[idx] = acc + pos[(row & 511) * 256 + d];
}

// ---------- processor embedding, bf16 out ----------
__global__ void pe_kernel(const float* __restrict__ sp, const float* __restrict__ av,
                          const float* __restrict__ g, const float* __restrict__ bb,
                          const float* __restrict__ W, const float* __restrict__ pb,
                          unsigned short* __restrict__ pe) {
  int idx = blockIdx.x * blockDim.x + threadIdx.x;  // B*M*P = 32768
  int bm = idx >> 6, p = idx & 63;
  float a0 = sp[bm] * 0.1f, a1 = av[bm];
  float mu = 0.5f * (a0 + a1);
  float d0 = a0 - mu, d1 = a1 - mu;
  float rr = rsqrtf(0.5f * (d0 * d0 + d1 * d1) + 1e-5f);
  float n0 = d0 * rr * g[0] + bb[0];
  float n1 = d1 * rr * g[1] + bb[1];
  float acc = pb[p] + n0 * W[p * 2] + n1 * W[p * 2 + 1];
  pe[idx] = f2bf(fmaxf(acc, 0.f));
}

// ---------- policy logits: wave per (b,n,m), relu(U+V).dot(w2) ----------
__global__ void logits_kernel(const float* __restrict__ U, const float* __restrict__ Vp,
                              const float* __restrict__ w2, const float* __restrict__ b2,
                              float* __restrict__ out) {
  int gw = (blockIdx.x * blockDim.x + threadIdx.x) >> 5;
  int lane = threadIdx.x & 31;
  int b = gw >> 15, rem = gw & 32767;
  int n = rem >> 6, m = rem & 63;
  const float* up = U + (size_t)(b * 512 + n) * 256;
  const float* vp = Vp + (size_t)(b * 64 + m) * 256;
  float acc = 0.f;
  for (int k = lane; k < 256; k += 32) {
    float hv = fmaxf(up[k] + vp[k], 0.f);
    acc += hv * w2[k];
  }
  for (int off = 16; off > 0; off >>= 1) acc += __shfl_xor(acc, off, 32);
  if (lane == 0) out[gw] = acc + b2[0];
}

// ---------- masked-mean graph embedding ----------
__global__ void ge_kernel(const float* __restrict__ x, const unsigned char* __restrict__ mask,
                          float* __restrict__ ge) {
  int idx = blockIdx.x * blockDim.x + threadIdx.x;  // B*D = 2048
  int b = idx >> 8, d = idx & 255;
  float sum = 0.f, cnt = 0.f;
  for (int n = 0; n < 512; ++n) {
    float mf = mask[b * 512 + n] ? 1.f : 0.f;
    sum += x[(size_t)(b * 512 + n) * 256 + d] * mf;
    cnt += mf;
  }
  ge[idx] = sum / fmaxf(cnt, 1.f);
}

__global__ void vh_kernel(const float* __restrict__ ge, const float* __restrict__ W,
                          const float* __restrict__ bias, float* __restrict__ vh) {
  int idx = blockIdx.x * blockDim.x + threadIdx.x;  // B*VH = 2048
  int b = idx >> 8, o_ = idx & 255;
  const float* g = ge + b * 256;
  const float* w = W + o_ * 256;
  float acc = bias[o_];
  for (int k = 0; k < 256; ++k) acc += g[k] * w[k];
  vh[idx] = fmaxf(acc, 0.f);
}

__global__ void final_kernel(const float* __restrict__ vh, const float* __restrict__ W2,
                             const float* __restrict__ b2, const float* __restrict__ lvp,
                             const float* __restrict__ lvv, float* __restrict__ out) {
  int t = threadIdx.x;
  if (t < 8) {
    const float* v = vh + t * 256;
    float acc = b2[0];
    for (int k = 0; k < 256; ++k) acc += v[k] * W2[k];
    out[262144 + t] = tanhf(acc);
  } else if (t == 8) out[262152] = lvp[0];
  else if (t == 9)   out[262153] = lvv[0];
}

extern "C" void kernel_launch(void* const* d_in, const int* in_sizes, int n_in,
                              void* d_out, int out_size, void* d_ws, size_t ws_size,
                              hipStream_t stream) {
  (void)in_sizes; (void)n_in; (void)out_size; (void)ws_size;
  const float* task_status = (const float*)d_in[0];
  const float* comp_costs  = (const float*)d_in[1];
  const float* comm_costs  = (const float*)d_in[2];
  const float* proc_speeds = (const float*)d_in[3];
  const float* proc_avail  = (const float*)d_in[4];
  const float* feat_ln_g   = (const float*)d_in[5];
  const float* feat_ln_b   = (const float*)d_in[6];
  const float* node_W      = (const float*)d_in[7];
  const float* node_b      = (const float*)d_in[8];
  const float* pos_emb     = (const float*)d_in[9];
  const float* edge_W      = (const float*)d_in[10];
  const float* Wqkv        = (const float*)d_in[11];
  const float* bqkv        = (const float*)d_in[12];
  const float* Wo          = (const float*)d_in[13];
  const float* bo          = (const float*)d_in[14];
  const float* ln1g        = (const float*)d_in[15];
  const float* ln1b        = (const float*)d_in[16];
  const float* ln2g        = (const float*)d_in[17];
  const float* ln2b        = (const float*)d_in[18];
  const float* ffW1        = (const float*)d_in[19];
  const float* ffb1        = (const float*)d_in[20];
  const float* ffW2        = (const float*)d_in[21];
  const float* ffb2        = (const float*)d_in[22];
  const float* proc_ln_g   = (const float*)d_in[23];
  const float* proc_ln_b   = (const float*)d_in[24];
  const float* proc_W      = (const float*)d_in[25];
  const float* proc_b      = (const float*)d_in[26];
  const float* pol_W1      = (const float*)d_in[27];
  const float* pol_b1      = (const float*)d_in[28];
  const float* pol_W2      = (const float*)d_in[29];
  const float* pol_b2      = (const float*)d_in[30];
  const float* val_W1      = (const float*)d_in[31];
  const float* val_b1      = (const float*)d_in[32];
  const float* val_W2      = (const float*)d_in[33];
  const float* val_b2      = (const float*)d_in[34];
  const float* log_var_pol = (const float*)d_in[35];
  const float* log_var_val = (const float*)d_in[36];
  const int*   adj         = (const int*)d_in[37];
  const unsigned char* task_mask = (const unsigned char*)d_in[38];
  float* out = (float*)d_out;

  // workspace bump allocator (256B aligned)
  char* base = (char*)d_ws;
  size_t off = 0;
  auto alloc = [&](size_t bytes) {
    void* p = base + off;
    off = (off + bytes + 255) & ~(size_t)255;
    return p;
  };
  float*          x       = (float*)alloc(4096u * 256 * 4);
  float*          U       = (float*)alloc(4096u * 256 * 4);
  float*          Vp      = (float*)alloc(512u * 256 * 4);
  float*          nfln    = (float*)alloc(4096u * 8 * 4);
  float*          ge      = (float*)alloc(8u * 256 * 4);
  float*          vh      = (float*)alloc(8u * 256 * 4);
  unsigned short* xn_bf   = (unsigned short*)alloc(4096u * 256 * 2);
  unsigned short* qkv_bf  = (unsigned short*)alloc(4096u * 768 * 2);
  unsigned short* o_bf    = (unsigned short*)alloc(4096u * 256 * 2);
  unsigned short* ffh_bf  = (unsigned short*)alloc(4096u * 1024 * 2);
  unsigned short* x_bf    = (unsigned short*)alloc(4096u * 256 * 2);
  unsigned short* pe_bf   = (unsigned short*)alloc(512u * 64 * 2);
  unsigned short* wqkv_bf = (unsigned short*)alloc(4u * 768 * 256 * 2);
  unsigned short* wo_bf   = (unsigned short*)alloc(4u * 256 * 256 * 2);
  unsigned short* ffw1_bf = (unsigned short*)alloc(4u * 1024 * 256 * 2);
  unsigned short* ffw2_bf = (unsigned short*)alloc(4u * 256 * 1024 * 2);
  unsigned short* pw1_bf  = (unsigned short*)alloc(256u * 320 * 2);

  // weight conversions (bf16, packed pairs)
  cvt_bf16_kernel<<<1536, 256, 0, stream>>>(Wqkv,   (unsigned int*)wqkv_bf, 393216);
  cvt_bf16_kernel<<<512,  256, 0, stream>>>(Wo,     (unsigned int*)wo_bf,   131072);
  cvt_bf16_kernel<<<2048, 256, 0, stream>>>(ffW1,   (unsigned int*)ffw1_bf, 524288);
  cvt_bf16_kernel<<<2048, 256, 0, stream>>>(ffW2,   (unsigned int*)ffw2_bf, 524288);
  cvt_bf16_kernel<<<160,  256, 0, stream>>>(pol_W1, (unsigned int*)pw1_bf,  40960);

  // input embedding
  nfln_kernel<<<32, 128, 0, stream>>>(task_status, comp_costs, feat_ln_g, feat_ln_b, nfln, 4096);
  nodeproj_kernel<<<4096, 256, 0, stream>>>(nfln, node_W, node_b, pos_emb, x);

  // transformer layers
  for (int l = 0; l < 4; ++l) {
    ln256_kernel<<<4096, 256, 0, stream>>>(x, ln1g + l * 256, ln1b + l * 256, xn_bf);
    gemm_bf16_kernel<<<dim3(12, 64), 256, 0, stream>>>(xn_bf, wqkv_bf + (size_t)l * 768 * 256,
        bqkv + l * 768, nullptr, nullptr, qkv_bf, 4096, 768, 256, 256, 0);
    attn_kernel<<<2048, 256, 0, stream>>>(qkv_bf, comm_costs, adj, task_mask, edge_W, o_bf);
    gemm_bf16_kernel<<<dim3(4, 64), 256, 0, stream>>>(o_bf, wo_bf + (size_t)l * 256 * 256,
        bo + l * 256, x, x, nullptr, 4096, 256, 256, 256, 0);
    ln256_kernel<<<4096, 256, 0, stream>>>(x, ln2g + l * 256, ln2b + l * 256, xn_bf);
    gemm_bf16_kernel<<<dim3(16, 64), 256, 0, stream>>>(xn_bf, ffw1_bf + (size_t)l * 1024 * 256,
        ffb1 + l * 1024, nullptr, nullptr, ffh_bf, 4096, 1024, 256, 256, 1);
    gemm_bf16_kernel<<<dim3(4, 64), 256, 0, stream>>>(ffh_bf, ffw2_bf + (size_t)l * 256 * 1024,
        ffb2 + l * 256, x, x, nullptr, 4096, 256, 1024, 1024, 0);
  }

  // processor embedding + factorized policy head
  pe_kernel<<<128, 256, 0, stream>>>(proc_speeds, proc_avail, proc_ln_g, proc_ln_b, proc_W, proc_b, pe_bf);
  cvt_bf16_kernel<<<2048, 256, 0, stream>>>(x, (unsigned int*)x_bf, 524288);
  gemm_bf16_kernel<<<dim3(4, 64), 256, 0, stream>>>(x_bf, pw1_bf, pol_b1, nullptr, U, nullptr,
      4096, 256, 256, 320, 0);
  gemm_bf16_kernel<<<dim3(4, 8), 256, 0, stream>>>(pe_bf, pw1_bf + 256, nullptr, nullptr, Vp, nullptr,
      512, 256, 64, 320, 0);
  logits_kernel<<<32768, 256, 0, stream>>>(U, Vp, pol_W2, pol_b2, out);

  // value head
  ge_kernel<<<8, 256, 0, stream>>>(x, task_mask, ge);
  vh_kernel<<<8, 256, 0, stream>>>(ge, val_W1, val_b1, vh);
  final_kernel<<<1, 32, 0, stream>>>(vh, val_W2, val_b2, log_var_pol, log_var_val, out);
}
